// Model_47760036331799
// MI455X (gfx1250) — matrix operand order repaired and checked by reading, not compiled, not run on hardware
//
#include <hip/hip_runtime.h>

// ---------------- Problem constants (fixed by reference) ----------------
#define EMBED   96
#define HEADS   8
#define HDW     12          // head dim
#define DD      32
#define HH      64
#define WW      64
#define TOK     64          // tokens per block (8 windows of 8 tokens)
#define THREADS 128         // 4 wave32 waves
#define PITCH   104         // padded row pitch (elements) for token-major LDS buffers (16B-aligned rows)

typedef _Float16 v16h __attribute__((ext_vector_type(16)));
typedef _Float16 v8h  __attribute__((ext_vector_type(8)));
typedef float    v8f  __attribute__((ext_vector_type(8)));

#ifndef __has_builtin
#define __has_builtin(x) 0
#endif
#if __has_builtin(__builtin_amdgcn_tensor_load_to_lds)
#define HAVE_TDM 1
#else
#define HAVE_TDM 0
#endif

// ---------------- LDS layout (byte offsets, all 16B aligned) ----------------
// raw buffers: [c][64] contiguous slab, exactly the TDM tile order [zz=c][z=d][y=h][x=w16]
#define OFF_RAWX 0                         // 96*64 f32  = 24576 B
#define OFF_RAWY (OFF_RAWX + 24576)        // 24576 B
#define OFF_AX   (OFF_RAWY + 24576)        // 64*104 f16 = 13312 B
#define OFF_AY   (OFF_AX   + 13312)        // 13312 B
#define OFF_WKV  (OFF_AY   + 13312)        // 192*104 f16 = 39936 B (transposed [N][K])
#define OFF_WQ   (OFF_WKV  + 39936)        //  96*104 f16 = 19968 B (transposed [N][K])
#define OFF_K    (OFF_WQ   + 19968)        // 64*104 f16 = 13312 B
#define OFF_V    (OFF_K    + 13312)        // 13312 B
#define OFF_Q    (OFF_V    + 13312)        // 13312 B
#define OFF_OUT  (OFF_Q    + 13312)        // 64*104 f32 = 26624 B
#define OFF_PAR  (OFF_OUT  + 26624)        // 864 f32 = 3456 B
#define SMEM_BYTES (OFF_PAR + 3456)        // = 205696 B (< 320KB WGP LDS)

// A fragment (16-bit, 16x32): lane 0-15 -> M=lane, K pairs {0..7}+kc*32 in v0-3,
// {16..23}+kc*32 in v4-7; lanes 16-31 same M, K offset +8.
__device__ __forceinline__ v16h frag_a(const _Float16* __restrict__ rowp, int kbase) {
    const v8h lo = *(const v8h*)(rowp + kbase);
    const v8h hi = *(const v8h*)(rowp + kbase + 16);
    v16h r;
#pragma unroll
    for (int i = 0; i < 8; ++i) { r[i] = lo[i]; r[i + 8] = hi[i]; }
    return r;
}

// B fragment (16-bit, 32x16): lane 0-15 -> N=lane, K {0..7} then {16..23}; lanes 16-31 K offset +8.
// Weights stored transposed [N][K] in LDS so this is two contiguous 16B reads.
__device__ __forceinline__ v16h frag_b(const _Float16* __restrict__ p) {
    const v8h lo = *(const v8h*)(p);
    const v8h hi = *(const v8h*)(p + 16);
    v16h r;
#pragma unroll
    for (int i = 0; i < 8; ++i) { r[i] = lo[i]; r[i + 8] = hi[i]; }
    return r;
}

#if HAVE_TDM
// Tensor Data Mover: DMA a 4-D tile [x=16 W elems][y=2 H rows][z=2 D rows][zz=96 channels]
// of the (b,c,D,H,W) f32 tensor into LDS, contiguous in tile order -> [c][d][h][w16].
// D# fields per CDNA5 ISA (count=1, type=2, data_size=4B).
__device__ __forceinline__ void tdm_load_slab(unsigned lds_off, unsigned long long gaddr) {
    typedef unsigned int v4u __attribute__((ext_vector_type(4)));
    typedef int          v8i __attribute__((ext_vector_type(8)));
    typedef int          v4i __attribute__((ext_vector_type(4)));
    v4u g0;
    g0[0] = 1u;                                                   // count=1, user mode
    g0[1] = lds_off;                                              // lds_addr (bytes)
    g0[2] = (unsigned)(gaddr & 0xffffffffull);                    // global_addr[31:0]
    g0[3] = (unsigned)((gaddr >> 32) & 0x1ffffffull) | (2u << 30); // addr[56:32] | type=2
    v8i g1;
    g1[0] = (2 << 16);          // workgroup_mask=0 (not in cluster), data_size=2 (4B)
    g1[1] = (64 << 16);         // tensor_dim0 (W)=64, low 16 bits at [31:16]
    g1[2] = (64 << 16);         // tensor_dim0 hi=0 | tensor_dim1 (H)=64 low16
    g1[3] = (16 << 16);         // tensor_dim1 hi=0 | tile_dim0=16
    g1[4] = 2 | (2 << 16);      // tile_dim1=2 (h rows), tile_dim2=2 (d rows)
    g1[5] = 64;                 // tensor_dim0_stride = 64 (W) lo32
    g1[6] = (4096 << 16);       // stride0 hi16=0 | tensor_dim1_stride=4096 (H*W) lo16
    g1[7] = 0;                  // stride1 hi32
    v4i g2;
    g2[0] = 32;                 // tensor_dim2 (D)
    g2[1] = 96;                 // tensor_dim3 (C), iterate_enable=0
    g2[2] = 131072;             // tensor_dim2_stride = D*H*W... channel stride (D? no: C stride) lo32
    g2[3] = (96 << 16);         // stride2 hi16=0 | tile_dim3=96 (channels)
    v4i g3 = (v4i){0, 0, 0, 0}; // dim3_stride / dim4 / tile_dim4 unused
#if __clang_major__ >= 23
    v8i g4 = (v8i){0, 0, 0, 0, 0, 0, 0, 0};
    __builtin_amdgcn_tensor_load_to_lds(g0, g1, g2, g3, g4, 0);
#else
    __builtin_amdgcn_tensor_load_to_lds(g0, g1, g2, g3, 0);
#endif
}
#endif

__global__ __launch_bounds__(THREADS)
void swin3d_cross_attn_fused(const float* __restrict__ x_in,
                             const float* __restrict__ y_in,
                             const float* __restrict__ ln_x_w, const float* __restrict__ ln_x_b,
                             const float* __restrict__ ln_y_w, const float* __restrict__ ln_y_b,
                             const float* __restrict__ kv_w,   const float* __restrict__ kv_b,
                             const float* __restrict__ q_w,    const float* __restrict__ q_b,
                             const float* __restrict__ ln_o_w, const float* __restrict__ ln_o_b,
                             float* __restrict__ out)
{
    extern __shared__ char smem[];
    float*    rawx = (float*)(smem + OFF_RAWX);
    float*    rawy = (float*)(smem + OFF_RAWY);
    _Float16* Ax   = (_Float16*)(smem + OFF_AX);
    _Float16* Ay   = (_Float16*)(smem + OFF_AY);
    _Float16* Wkv  = (_Float16*)(smem + OFF_WKV);
    _Float16* Wq   = (_Float16*)(smem + OFF_WQ);
    _Float16* Kb   = (_Float16*)(smem + OFF_K);
    _Float16* Vb   = (_Float16*)(smem + OFF_V);
    _Float16* Qb   = (_Float16*)(smem + OFF_Q);
    float*    Ob   = (float*)(smem + OFF_OUT);
    float*    par  = (float*)(smem + OFF_PAR);
    // par: [0)lnxw [96)lnxb [192)lnyw [288)lnyb [384)lnow [480)lnob [576)kvb(192) [768)qb(96)

    const int tid = threadIdx.x;
    const int bw  = blockIdx.x;        // W chunk of 16 elements (8 windows)
    const int h2  = blockIdx.y;        // H window row
    const int d2  = blockIdx.z & 15;   // D window row
    const int bb  = blockIdx.z >> 4;   // batch

    // ---------------- Phase 0: input slabs + weights -> LDS ----------------
#if HAVE_TDM
    if (tid < 32) {   // one TDM issue per block (per-wave DMA op; EXEC ignored)
        const size_t ebase = ((((size_t)bb * EMBED) * DD + d2 * 2) * HH + h2 * 2) * (size_t)WW
                             + (size_t)bw * 16;
        tdm_load_slab((unsigned)(uintptr_t)(void*)rawx,
                      (unsigned long long)(uintptr_t)(x_in + ebase));
        tdm_load_slab((unsigned)(uintptr_t)(void*)rawy,
                      (unsigned long long)(uintptr_t)(y_in + ebase));
    }
#else
    for (int j = tid; j < EMBED * TOK; j += THREADS) {
        const int c = j >> 6, rem = j & 63;
        const int din = rem >> 5, hin = (rem >> 4) & 1, wo = rem & 15;
        const int d = d2 * 2 + din, h = h2 * 2 + hin, w = bw * 16 + wo;
        const size_t g = ((((size_t)bb * EMBED + c) * DD + d) * HH + h) * (size_t)WW + w;
        rawx[j] = x_in[g];
        rawy[j] = y_in[g];
    }
#endif
    // Weights, transposed to [N][K] f16 for contiguous B-fragment reads (overlaps TDM).
    for (int j = tid; j < EMBED * 2 * EMBED; j += THREADS) {
        const int k = j / (2 * EMBED), n = j % (2 * EMBED);
        Wkv[n * PITCH + k] = (_Float16)kv_w[j];
    }
    for (int j = tid; j < EMBED * EMBED; j += THREADS) {
        const int k = j / EMBED, n = j % EMBED;
        Wq[n * PITCH + k] = (_Float16)q_w[j];
    }
    if (tid < EMBED) {
        par[tid]       = ln_x_w[tid]; par[96  + tid] = ln_x_b[tid];
        par[192 + tid] = ln_y_w[tid]; par[288 + tid] = ln_y_b[tid];
        par[384 + tid] = ln_o_w[tid]; par[480 + tid] = ln_o_b[tid];
        par[576 + tid] = kv_b[tid];   par[672 + tid] = kv_b[96 + tid];
        par[768 + tid] = q_b[tid];
    }
#if HAVE_TDM
    if (tid < 32) __builtin_amdgcn_s_wait_tensorcnt(0);
#endif
    __syncthreads();

    // ---------------- Phase 1: per-token LayerNorm -> f16 A matrices ----------------
    // raw layout is [c][din][hin][wo]; token t -> offset din*32 + hin*16 + ww*2 + w_in
    {
        const int t = tid & 63;
        const bool isx = tid < 64;
        const float* raw = isx ? rawx : rawy;
        const float* lw  = isx ? (par + 0)  : (par + 192);
        const float* lb  = isx ? (par + 96) : (par + 288);
        _Float16* A      = isx ? Ax : Ay;
        const int toff = ((t >> 2) & 1) * 32 + ((t >> 1) & 1) * 16 + (t >> 3) * 2 + (t & 1);
        float s = 0.f;
#pragma unroll 8
        for (int c = 0; c < EMBED; ++c) s += raw[c * 64 + toff];
        const float mu = s * (1.0f / EMBED);
        float vs = 0.f;
#pragma unroll 8
        for (int c = 0; c < EMBED; ++c) { const float d0 = raw[c * 64 + toff] - mu; vs += d0 * d0; }
        const float rs = rsqrtf(vs * (1.0f / EMBED) + 1e-5f);
#pragma unroll 8
        for (int c = 0; c < EMBED; ++c)
            A[t * PITCH + c] = (_Float16)((raw[c * 64 + toff] - mu) * rs * lw[c] + lb[c]);
    }
    __syncthreads();

    // ---------------- Phase 2: WMMA projections (kv from x, q from y) ----------------
    {
        const int wave  = tid >> 5;             // M tile (16 tokens each)
        const int lane  = tid & 31;
        const int mrow  = wave * 16 + (lane & 15);
        const int koffA = (lane & 16) ? 8 : 0;
        const int koffB = (lane & 16) ? 8 : 0;

        const v16h ax0 = frag_a(&Ax[mrow * PITCH], 0  + koffA);
        const v16h ax1 = frag_a(&Ax[mrow * PITCH], 32 + koffA);
        const v16h ax2 = frag_a(&Ax[mrow * PITCH], 64 + koffA);
        const v16h ay0 = frag_a(&Ay[mrow * PITCH], 0  + koffA);
        const v16h ay1 = frag_a(&Ay[mrow * PITCH], 32 + koffA);
        const v16h ay2 = frag_a(&Ay[mrow * PITCH], 64 + koffA);

        const int ncol  = lane & 15;
        const int rbase = wave * 16 + ((lane & 16) ? 8 : 0);

        // K/V projection: 12 N tiles over 192 output columns, K=96 in 3 chunks.
        for (int nt = 0; nt < 12; ++nt) {
            const _Float16* bp = &Wkv[(nt * 16 + ncol) * PITCH + koffB];
            v8f acc = {};
            acc = __builtin_amdgcn_wmma_f32_16x16x32_f16(false, ax0, false, frag_b(bp),      (short)0, acc, false, false);
            acc = __builtin_amdgcn_wmma_f32_16x16x32_f16(false, ax1, false, frag_b(bp + 32), (short)0, acc, false, false);
            acc = __builtin_amdgcn_wmma_f32_16x16x32_f16(false, ax2, false, frag_b(bp + 64), (short)0, acc, false, false);
            const int n = nt * 16 + ncol;
            const float bias = par[576 + n];
            _Float16* dst = (n < EMBED) ? &Kb[n] : &Vb[n - EMBED];
#pragma unroll
            for (int i = 0; i < 8; ++i)
                dst[(rbase + i) * PITCH] = (_Float16)(acc[i] + bias);
        }
        // Q projection: 6 N tiles over 96 columns; scale by hd^-0.5 after bias.
        for (int nt = 0; nt < 6; ++nt) {
            const _Float16* bp = &Wq[(nt * 16 + ncol) * PITCH + koffB];
            v8f acc = {};
            acc = __builtin_amdgcn_wmma_f32_16x16x32_f16(false, ay0, false, frag_b(bp),      (short)0, acc, false, false);
            acc = __builtin_amdgcn_wmma_f32_16x16x32_f16(false, ay1, false, frag_b(bp + 32), (short)0, acc, false, false);
            acc = __builtin_amdgcn_wmma_f32_16x16x32_f16(false, ay2, false, frag_b(bp + 64), (short)0, acc, false, false);
            const int n = nt * 16 + ncol;
            const float bias = par[768 + n];
#pragma unroll
            for (int i = 0; i < 8; ++i)
                Qb[(rbase + i) * PITCH + n] = (_Float16)((acc[i] + bias) * 0.28867513459481287f);
        }
    }
    __syncthreads();

    // ---------------- Phase 3: per-(window, head) attention (8x8, hd=12) ----------------
    if (tid < 64) {
        const int wi   = tid >> 3;
        const int hoff = (tid & 7) * HDW;
        const int tb   = wi * 8;
        for (int qi = 0; qi < 8; ++qi) {
            const _Float16* qr = &Qb[(tb + qi) * PITCH + hoff];
            float lg[8], mx = -1e30f;
#pragma unroll
            for (int kj = 0; kj < 8; ++kj) {
                const _Float16* kr = &Kb[(tb + kj) * PITCH + hoff];
                float s = 0.f;
#pragma unroll
                for (int e = 0; e < HDW; ++e) s += (float)qr[e] * (float)kr[e];
                lg[kj] = s; mx = fmaxf(mx, s);
            }
            float den = 0.f;
#pragma unroll
            for (int kj = 0; kj < 8; ++kj) { lg[kj] = __expf(lg[kj] - mx); den += lg[kj]; }
            const float inv = 1.0f / den;
            float o[HDW];
#pragma unroll
            for (int e = 0; e < HDW; ++e) o[e] = 0.f;
#pragma unroll
            for (int kj = 0; kj < 8; ++kj) {
                const float p = lg[kj] * inv;
                const _Float16* vr = &Vb[(tb + kj) * PITCH + hoff];
#pragma unroll
                for (int e = 0; e < HDW; ++e) o[e] += p * (float)vr[e];
            }
#pragma unroll
            for (int e = 0; e < HDW; ++e) Ob[(tb + qi) * PITCH + hoff + e] = o[e];
        }
    }
    __syncthreads();

    // ---------------- Phase 4: output LayerNorm (in place) ----------------
    if (tid < 64) {
        float s = 0.f;
#pragma unroll 8
        for (int c = 0; c < EMBED; ++c) s += Ob[tid * PITCH + c];
        const float mu = s * (1.0f / EMBED);
        float vs = 0.f;
#pragma unroll 8
        for (int c = 0; c < EMBED; ++c) { const float d0 = Ob[tid * PITCH + c] - mu; vs += d0 * d0; }
        const float rs = rsqrtf(vs * (1.0f / EMBED) + 1e-5f);
#pragma unroll 8
        for (int c = 0; c < EMBED; ++c)
            Ob[tid * PITCH + c] = (Ob[tid * PITCH + c] - mu) * rs * par[384 + c] + par[480 + c];
    }
    __syncthreads();

    // ---------------- Phase 5: coalesced channel-major store ----------------
    for (int j = tid; j < EMBED * TOK; j += THREADS) {
        const int c = j >> 6, rem = j & 63;
        const int r = rem >> 4, wo = rem & 15;
        const int d = d2 * 2 + (r >> 1);
        const int h = h2 * 2 + (r & 1);
        const int w = bw * 16 + wo;
        const size_t g = ((((size_t)bb * EMBED + c) * DD + d) * HH + h) * (size_t)WW + w;
        const int tok = (wo >> 1) * 8 + (r >> 1) * 4 + (r & 1) * 2 + (wo & 1);
        out[g] = Ob[tok * PITCH + c];
    }
}

extern "C" void kernel_launch(void* const* d_in, const int* in_sizes, int n_in,
                              void* d_out, int out_size, void* d_ws, size_t ws_size,
                              hipStream_t stream) {
    (void)n_in; (void)out_size; (void)d_ws; (void)ws_size;
    const float* x_in   = (const float*)d_in[0];
    const float* y_in   = (const float*)d_in[1];
    const float* ln_x_w = (const float*)d_in[2];
    const float* ln_x_b = (const float*)d_in[3];
    const float* ln_y_w = (const float*)d_in[4];
    const float* ln_y_b = (const float*)d_in[5];
    const float* kv_w   = (const float*)d_in[6];
    const float* kv_b   = (const float*)d_in[7];
    const float* q_w    = (const float*)d_in[8];
    const float* q_b    = (const float*)d_in[9];
    const float* ln_o_w = (const float*)d_in[10];
    const float* ln_o_b = (const float*)d_in[11];
    float* out = (float*)d_out;

    const int b = in_sizes[0] / (EMBED * DD * HH * WW);   // = 2
    dim3 grid(WW / 16, HH / 2, (DD / 2) * b);             // (4, 32, 32)
    swin3d_cross_attn_fused<<<grid, dim3(THREADS), SMEM_BYTES, stream>>>(
        x_in, y_in, ln_x_w, ln_x_b, ln_y_w, ln_y_b,
        kv_w, kv_b, q_w, q_b, ln_o_w, ln_o_b, out);
}
